// CrossAttention_61512521614018
// MI455X (gfx1250) — matrix operand order repaired and checked by reading, not compiled, not run on hardware
//
#include <hip/hip_runtime.h>
#include <hip/hip_bf16.h>

// ---------------------------------------------------------------------------
// CrossAttention on MI455X (gfx1250): bf16 WMMA pipeline
//   1) convert dec/enc fp32->bf16; transpose+convert weights to [N][K] bf16
//   2) Q/K/V projection GEMMs via v_wmma_f32_16x16x32_bf16 (head-split layouts,
//      V stored transposed [B][H][64][Sk], 1/8 scale folded into Q)
//   3) flash-attention: per-wave 16-query tile, 32-key blocks, online softmax,
//      P transposed through LDS (s_wait_dscnt) for the PV WMMA
//   4) output projection GEMM -> fp32 d_out
// ---------------------------------------------------------------------------

typedef __bf16 bf16;
typedef __attribute__((ext_vector_type(16))) __bf16 v16bf;
typedef __attribute__((ext_vector_type(8)))  float  v8f;
typedef __attribute__((ext_vector_type(4)))  int    v4i;

union BFragU { v16bf v; v4i q[2]; };

// A-fragment (16x32, M x K): lane m = lane&15, kb = (lane>=16)*8,
// elems 0..7 = K kb..kb+7, elems 8..15 = K 16+kb..16+kb+7  -> two 16B loads
__device__ __forceinline__ v16bf load_frag_2x16B(const bf16* p0, const bf16* p1) {
  BFragU f;
  f.q[0] = *(const v4i*)p0;
  f.q[1] = *(const v4i*)p1;
  return f.v;
}
// B-fragment (32x16, K x N): lane n = lane&15, kb = (lane>=16)*8,
// elems 0..7 = K kb..kb+7, elems 8..15 = K 16+kb..16+kb+7  -> two 16B loads
__device__ __forceinline__ v16bf load_frag_32B(const bf16* p) {
  BFragU f;
  f.q[0] = *(const v4i*)p;
  f.q[1] = *(const v4i*)(p + 16);
  return f.v;
}

__device__ __forceinline__ v8f wmma_bf16(v16bf a, v16bf b, v8f c) {
  return __builtin_amdgcn_wmma_f32_16x16x32_bf16(
      /*neg_a=*/false, a, /*neg_b=*/false, b,
      /*c_mod=*/(short)0, c, /*reuse_a=*/false, /*reuse_b=*/false);
}

__device__ __forceinline__ float hmax16(float v) {
  v = fmaxf(v, __shfl_xor(v, 1));
  v = fmaxf(v, __shfl_xor(v, 2));
  v = fmaxf(v, __shfl_xor(v, 4));
  v = fmaxf(v, __shfl_xor(v, 8));
  return v;
}
__device__ __forceinline__ float hsum16(float v) {
  v += __shfl_xor(v, 1);
  v += __shfl_xor(v, 2);
  v += __shfl_xor(v, 4);
  v += __shfl_xor(v, 8);
  return v;
}

// ------------------------------- converters --------------------------------

__global__ void ca_cvt_f32_bf16(const float* __restrict__ in,
                                bf16* __restrict__ out, int n) {
  int i = blockIdx.x * blockDim.x + threadIdx.x;
  if (i < n) out[i] = (bf16)in[i];
}

// in: [1024][1024] fp32 row-major (K x N) -> out: [N][K] bf16
__global__ void ca_transpose_bf16(const float* __restrict__ in,
                                  bf16* __restrict__ out) {
  int i = blockIdx.x * blockDim.x + threadIdx.x;
  int k = i >> 10, n = i & 1023;
  out[n * 1024 + k] = (bf16)in[i];
}

// ---------------------------- projection GEMM ------------------------------
// C[M,1024] = A_bf[M,1024] @ W  (W given pre-transposed as Wt[N=1024][K=1024])
// Epilogue: +bias, *scale, scatter to head-split layout:
//   mode 0: out[((b*16+h)*S + s)*64 + d]       (Q, K)
//   mode 2: out[((b*16+h)*64 + d)*S + s]       (V transposed)
// Block: 256 thr = 8 waves; wave tile 32x32 (2x2 accumulators); grid (M/128, 16)
__global__ void __launch_bounds__(256)
ca_gemm_proj(const bf16* __restrict__ A, const bf16* __restrict__ Wt,
             const float* __restrict__ bias, bf16* __restrict__ out,
             int S, int mode, float scale) {
  const int lane = threadIdx.x & 31;
  const int half = lane >> 4;
  const int ln   = lane & 15;
  const int wave = threadIdx.x >> 5;
  const int m_base = blockIdx.x * 128 + (wave & 3) * 32;
  const int n_base = blockIdx.y * 64  + (wave >> 2) * 32;
  const int K = 1024;

  const bf16* arow0 = A  + (size_t)(m_base + ln)      * K;
  const bf16* arow1 = A  + (size_t)(m_base + 16 + ln) * K;
  const bf16* wrow0 = Wt + (size_t)(n_base + ln)      * K;
  const bf16* wrow1 = Wt + (size_t)(n_base + 16 + ln) * K;
  const int kba = half * 8;
  const int kbb = half * 8;

  v8f acc[2][2] = {};
#pragma unroll 4
  for (int k0 = 0; k0 < K; k0 += 32) {
    v16bf a0 = load_frag_2x16B(arow0 + k0 + kba, arow0 + k0 + 16 + kba);
    v16bf a1 = load_frag_2x16B(arow1 + k0 + kba, arow1 + k0 + 16 + kba);
    v16bf b0 = load_frag_32B(wrow0 + k0 + kbb);
    v16bf b1 = load_frag_32B(wrow1 + k0 + kbb);
    acc[0][0] = wmma_bf16(a0, b0, acc[0][0]);
    acc[0][1] = wmma_bf16(a0, b1, acc[0][1]);
    acc[1][0] = wmma_bf16(a1, b0, acc[1][0]);
    acc[1][1] = wmma_bf16(a1, b1, acc[1][1]);
  }

#pragma unroll
  for (int j = 0; j < 2; ++j) {
    const int gn = n_base + j * 16 + ln;
    const float bv = bias[gn];
    const int hh = gn >> 6, d = gn & 63;
#pragma unroll
    for (int i = 0; i < 2; ++i) {
#pragma unroll
      for (int r = 0; r < 8; ++r) {
        const int gm = m_base + i * 16 + r + half * 8;
        const int bidx = gm / S;
        const int s = gm - bidx * S;
        const float v = (acc[i][j][r] + bv) * scale;
        size_t oidx;
        if (mode == 2) oidx = ((size_t)(bidx * 16 + hh) * 64 + d) * (size_t)S + s;
        else           oidx = ((size_t)(bidx * 16 + hh) * (size_t)S + s) * 64 + d;
        out[oidx] = (bf16)v;
      }
    }
  }
}

// ----------------------------- output GEMM ---------------------------------
// d_out[4096,1024] fp32 = attn_bf[4096,1024] @ o_w + o_b  (o_w as Wt[N][K])
__global__ void __launch_bounds__(256)
ca_gemm_out(const bf16* __restrict__ A, const bf16* __restrict__ Wt,
            const float* __restrict__ bias, float* __restrict__ out) {
  const int lane = threadIdx.x & 31;
  const int half = lane >> 4;
  const int ln   = lane & 15;
  const int wave = threadIdx.x >> 5;
  const int m_base = blockIdx.x * 128 + (wave & 3) * 32;
  const int n_base = blockIdx.y * 64  + (wave >> 2) * 32;
  const int K = 1024;

  const bf16* arow0 = A  + (size_t)(m_base + ln)      * K;
  const bf16* arow1 = A  + (size_t)(m_base + 16 + ln) * K;
  const bf16* wrow0 = Wt + (size_t)(n_base + ln)      * K;
  const bf16* wrow1 = Wt + (size_t)(n_base + 16 + ln) * K;
  const int kba = half * 8;
  const int kbb = half * 8;

  v8f acc[2][2] = {};
#pragma unroll 4
  for (int k0 = 0; k0 < K; k0 += 32) {
    v16bf a0 = load_frag_2x16B(arow0 + k0 + kba, arow0 + k0 + 16 + kba);
    v16bf a1 = load_frag_2x16B(arow1 + k0 + kba, arow1 + k0 + 16 + kba);
    v16bf b0 = load_frag_32B(wrow0 + k0 + kbb);
    v16bf b1 = load_frag_32B(wrow1 + k0 + kbb);
    acc[0][0] = wmma_bf16(a0, b0, acc[0][0]);
    acc[0][1] = wmma_bf16(a0, b1, acc[0][1]);
    acc[1][0] = wmma_bf16(a1, b0, acc[1][0]);
    acc[1][1] = wmma_bf16(a1, b1, acc[1][1]);
  }

#pragma unroll
  for (int j = 0; j < 2; ++j) {
    const int gn = n_base + j * 16 + ln;
    const float bv = bias[gn];
#pragma unroll
    for (int i = 0; i < 2; ++i) {
#pragma unroll
      for (int r = 0; r < 8; ++r) {
        const int gm = m_base + i * 16 + r + half * 8;
        out[(size_t)gm * 1024 + gn] = acc[i][j][r] + bv;
      }
    }
  }
}

// --------------------------- flash attention -------------------------------
// One wave per (b, h, 16-query tile). Q pre-scaled by 1/8.
// Q: [B][H][1024][64] bf16, K: [B][H][2048][64] bf16, Vt: [B][H][64][2048] bf16
// mask: [B][2048] bytes (nonzero = masked). Output attn: [B][1024][1024] bf16.
__global__ void __launch_bounds__(128)
ca_flash_attn(const bf16* __restrict__ Q, const bf16* __restrict__ Kt,
              const bf16* __restrict__ Vt, const unsigned char* __restrict__ mask,
              bf16* __restrict__ Oout) {
  __shared__ bf16 plds[4][16 * 32];   // per-wave P staging tile (1 KB each)

  const int lane  = threadIdx.x & 31;
  const int half  = lane >> 4;
  const int ln    = lane & 15;
  const int wslot = threadIdx.x >> 5;
  const int gw = blockIdx.x * 4 + wslot;
  const int qt = gw & 63;
  const int h  = (gw >> 6) & 15;
  const int b  = gw >> 10;

  const bf16* Qp = Q  + ((size_t)(b * 16 + h) * 1024 + qt * 16) * 64;
  const bf16* Kp = Kt + (size_t)(b * 16 + h) * 2048 * 64;
  const bf16* Vp = Vt + (size_t)(b * 16 + h) * 64 * 2048;
  const unsigned char* mp = mask + b * 2048;

  // Q tile A-fragments: d-chunks [0,32) and [32,64)
  const bf16* qrow = Qp + (size_t)ln * 64;
  const v16bf aq0 = load_frag_2x16B(qrow +      half * 8, qrow + 16 + half * 8);
  const v16bf aq1 = load_frag_2x16B(qrow + 32 + half * 8, qrow + 48 + half * 8);

  v8f oacc[4] = {};
  float m_r[8], l_r[8];
#pragma unroll
  for (int r = 0; r < 8; ++r) { m_r[r] = 0.0f; l_r[r] = 0.0f; }

  bf16* pl = plds[wslot];

  for (int key0 = 0; key0 < 2048; key0 += 32) {
    // ---- S = Q @ K^T for 32 keys (two 16-key column tiles) ----
    const bf16* krow0 = Kp + (size_t)(key0 + ln)      * 64;
    const bf16* krow1 = Kp + (size_t)(key0 + 16 + ln) * 64;
    v16bf bk0c0 = load_frag_32B(krow0 +      half * 8);
    v16bf bk0c1 = load_frag_32B(krow0 + 32 + half * 8);
    v16bf bk1c0 = load_frag_32B(krow1 +      half * 8);
    v16bf bk1c1 = load_frag_32B(krow1 + 32 + half * 8);
    v8f s0 = {}, s1 = {};
    s0 = wmma_bf16(aq0, bk0c0, s0);
    s0 = wmma_bf16(aq1, bk0c1, s0);
    s1 = wmma_bf16(aq0, bk1c0, s1);
    s1 = wmma_bf16(aq1, bk1c1, s1);

    const bool mk0 = mp[key0 + ln] != 0;
    const bool mk1 = mp[key0 + 16 + ln] != 0;

    // ---- online softmax (per row r; replicated across each 16-lane half) ----
    float p0[8], p1[8];
#pragma unroll
    for (int r = 0; r < 8; ++r) {
      const float v0 = mk0 ? -1e30f : s0[r];
      const float v1 = mk1 ? -1e30f : s1[r];
      const float bmax = hmax16(fmaxf(v0, v1));
      const float mnew = fmaxf(m_r[r], bmax);
      const float corr = __expf(m_r[r] - mnew);
      const float e0 = __expf(v0 - mnew);
      const float e1 = __expf(v1 - mnew);
      const float bsum = hsum16(e0 + e1);
      l_r[r] = l_r[r] * corr + bsum;
      m_r[r] = mnew;
#pragma unroll
      for (int t = 0; t < 4; ++t) oacc[t][r] *= corr;
      p0[r] = e0;
      p1[r] = e1;
    }

    // ---- transpose P (C-layout) -> A-layout through per-wave LDS tile ----
#pragma unroll
    for (int r = 0; r < 8; ++r) {
      const int row = r + half * 8;
      pl[row * 32 + ln]      = (bf16)p0[r];
      pl[row * 32 + 16 + ln] = (bf16)p1[r];
    }
    asm volatile("s_wait_dscnt 0x0" ::: "memory");
    const v16bf ap = load_frag_2x16B(pl + ln * 32 + half * 8,
                                     pl + ln * 32 + 16 + half * 8);

    // ---- O += P @ V (Vt rows are contiguous over keys) ----
#pragma unroll
    for (int t = 0; t < 4; ++t) {
      const v16bf bv =
          load_frag_32B(Vp + (size_t)(t * 16 + ln) * 2048 + key0 + half * 8);
      oacc[t] = wmma_bf16(ap, bv, oacc[t]);
    }
  }

  // ---- normalize and store: attn[b][row][h*64 + d] ----
#pragma unroll
  for (int r = 0; r < 8; ++r) {
    const float inv = 1.0f / l_r[r];
    const int row = qt * 16 + r + half * 8;
    bf16* orow = Oout + ((size_t)(b * 1024 + row)) * 1024 + h * 64;
#pragma unroll
    for (int t = 0; t < 4; ++t)
      orow[t * 16 + ln] = (bf16)(oacc[t][r] * inv);
  }
}

// ------------------------------- launcher ----------------------------------

extern "C" void kernel_launch(void* const* d_in, const int* in_sizes, int n_in,
                              void* d_out, int out_size, void* d_ws, size_t ws_size,
                              hipStream_t stream) {
  const float*         enc      = (const float*)d_in[0];          // [4,2048,1024]
  const unsigned char* enc_mask = (const unsigned char*)d_in[1];  // [4,1,1,2048] bool
  const float*         dec      = (const float*)d_in[2];          // [4,1024,1024]
  const float* q_w = (const float*)d_in[3];
  const float* q_b = (const float*)d_in[4];
  const float* k_w = (const float*)d_in[5];
  const float* k_b = (const float*)d_in[6];
  const float* v_w = (const float*)d_in[7];
  const float* v_b = (const float*)d_in[8];
  const float* o_w = (const float*)d_in[9];
  const float* o_b = (const float*)d_in[10];
  float* out = (float*)d_out;

  // workspace carve-up (bf16 elements; total 41,943,040 elems = 80 MiB)
  bf16* ws     = (bf16*)d_ws;
  bf16* dec_bf = ws;                    // 4,194,304
  bf16* enc_bf = dec_bf + 4194304;      // 8,388,608
  bf16* qwt    = enc_bf + 8388608;      // 1,048,576 each
  bf16* kwt    = qwt + 1048576;
  bf16* vwt    = kwt + 1048576;
  bf16* owt    = vwt + 1048576;
  bf16* Qb     = owt + 1048576;         // 4,194,304  [B][H][1024][64]
  bf16* Kb     = Qb  + 4194304;         // 8,388,608  [B][H][2048][64]
  bf16* Vtb    = Kb  + 8388608;         // 8,388,608  [B][H][64][2048]
  bf16* attn   = Vtb + 8388608;         // 4,194,304  [B][1024][1024]

  // 1) conversions
  ca_cvt_f32_bf16<<<4194304 / 256, 256, 0, stream>>>(dec, dec_bf, 4194304);
  ca_cvt_f32_bf16<<<8388608 / 256, 256, 0, stream>>>(enc, enc_bf, 8388608);
  ca_transpose_bf16<<<4096, 256, 0, stream>>>(q_w, qwt);
  ca_transpose_bf16<<<4096, 256, 0, stream>>>(k_w, kwt);
  ca_transpose_bf16<<<4096, 256, 0, stream>>>(v_w, vwt);
  ca_transpose_bf16<<<4096, 256, 0, stream>>>(o_w, owt);

  // 2) projections (1/sqrt(64)=0.125 folded exactly into Q)
  ca_gemm_proj<<<dim3(32, 16), 256, 0, stream>>>(dec_bf, qwt, q_b, Qb,  1024, 0, 0.125f);
  ca_gemm_proj<<<dim3(64, 16), 256, 0, stream>>>(enc_bf, kwt, k_b, Kb,  2048, 0, 1.0f);
  ca_gemm_proj<<<dim3(64, 16), 256, 0, stream>>>(enc_bf, vwt, v_b, Vtb, 2048, 2, 1.0f);

  // 3) attention: 4096 (b,h,qtile) waves, 4 waves/block
  ca_flash_attn<<<1024, 128, 0, stream>>>(Qb, Kb, Vtb, enc_mask, attn);

  // 4) output projection -> fp32
  ca_gemm_out<<<dim3(32, 16), 256, 0, stream>>>(attn, owt, o_b, out);
}
